// RiboPolisherMamba_2697239462360
// MI455X (gfx1250) — matrix operand order repaired and checked by reading, not compiled, not run on hardware
//
#include <hip/hip_runtime.h>
#include <hip/hip_bf16.h>
#include <math.h>

// ---------------- model dims (compile-time) ----------------
#define B_    2
#define L_    4096
#define DM_   192
#define NL_   4
#define DS_   16
#define DC_   4
#define DI_   384          // EXP * DM
#define DTR_  12
#define NXP_  44           // DTR + 2*DS
#define NPAD_ 64           // x-proj N padded to WMMA-friendly width
#define M_    (B_ * L_)    // 8192 rows
#define D2I_  (2 * DI_)    // 768

typedef __attribute__((ext_vector_type(16))) _Float16 v16h;
typedef __attribute__((ext_vector_type(8)))  _Float16 v8h;
typedef __attribute__((ext_vector_type(8)))  float    v8f;
typedef __attribute__((ext_vector_type(4)))  unsigned v4u;
typedef __attribute__((ext_vector_type(8)))  unsigned v8u;

// ---------------- small helpers ----------------
__device__ __forceinline__ float silu_f(float x) { return x / (1.0f + expf(-x)); }

// ---------------- weight prep: fp32 -> fp16 ----------------
__global__ void f32tof16_kernel(const float* __restrict__ s, _Float16* __restrict__ d, int n) {
    int i = blockIdx.x * blockDim.x + threadIdx.x;
    if (i < n) d[i] = (_Float16)s[i];
}

// xproj_w (NL,44,384) -> (NL,64,384) f16, rows 44..63 zero
__global__ void pad_xproj_kernel(const float* __restrict__ w, _Float16* __restrict__ d) {
    int i = blockIdx.x * blockDim.x + threadIdx.x;
    if (i >= NL_ * NPAD_ * DI_) return;
    int k = i % DI_;
    int r = (i / DI_) % NPAD_;
    int l = i / (DI_ * NPAD_);
    d[i] = (r < NXP_) ? (_Float16)w[((size_t)l * NXP_ + r) * DI_ + k] : (_Float16)0.0f;
}

// ---------------- embedding ----------------
__global__ void embed_kernel(const int* __restrict__ ids, const float* __restrict__ simf,
                             const int* __restrict__ ab, const int* __restrict__ pb,
                             const int* __restrict__ bk,
                             const float* __restrict__ cod_emb, const float* __restrict__ sim_w,
                             const float* __restrict__ sim_b, const float* __restrict__ aw,
                             const float* __restrict__ pw, const float* __restrict__ bw,
                             float* __restrict__ x) {
    int idx = blockIdx.x * blockDim.x + threadIdx.x;
    if (idx >= M_ * DM_) return;
    int row = idx / DM_, d = idx % DM_;
    int a = ab[row]; a = a < 0 ? 0 : (a > 3 ? 3 : a);
    int p = pb[row]; p = p < 0 ? 0 : (p > 3 ? 3 : p);
    int u = bk[row]; u = u < 0 ? 0 : (u > 2 ? 2 : u);
    x[idx] = cod_emb[(size_t)ids[row] * DM_ + d] + simf[row] * sim_w[d] + sim_b[d]
           + aw[a * DM_ + d] + pw[p * DM_ + d] + bw[u * DM_ + d];
}

// ---------------- layernorm (one wave32 per row, shfl reduce) ----------------
__global__ void ln_kernel(const float* __restrict__ x, const float* __restrict__ g,
                          const float* __restrict__ b, _Float16* __restrict__ h) {
    int wave = threadIdx.x >> 5, lane = threadIdx.x & 31;
    int row = blockIdx.x * (blockDim.x >> 5) + wave;
    if (row >= M_) return;
    const float* xr = x + (size_t)row * DM_;
    float v[6], s = 0.f, s2 = 0.f;
#pragma unroll
    for (int i = 0; i < 6; ++i) { float t = xr[lane + 32 * i]; v[i] = t; s += t; s2 += t * t; }
#pragma unroll
    for (int off = 16; off > 0; off >>= 1) {
        s  += __shfl_xor(s,  off, 32);
        s2 += __shfl_xor(s2, off, 32);
    }
    float m   = s / (float)DM_;
    float var = s2 / (float)DM_ - m * m;
    float r   = rsqrtf(var + 1e-5f);
    _Float16* hr = h + (size_t)row * DM_;
#pragma unroll
    for (int i = 0; i < 6; ++i) {
        int d = lane + 32 * i;
        hr[d] = (_Float16)((v[i] - m) * r * g[d] + b[d]);
    }
}

// ---------------- WMMA GEMM:  C(MxN) = A(MxK) * W(NxK)^T [+bias][+addsrc] ----------------
// Block = 8 waves = 8 M-tiles (128 rows) x one 64-wide N tile.
// The weight tile W[n0:n0+64, 0:K] (f16, <=48KB) is DMA'd into LDS once per block by the
// Tensor Data Mover (tensor_load_to_lds, TENSORcnt), then every wave reads its B fragments
// from LDS for the whole K loop. A fragments stream from global, double-buffered so the
// next K-step's loads overlap the current step's 4 WMMAs.
__global__ void wmma_gemm_kernel(const _Float16* __restrict__ A, const _Float16* __restrict__ W,
                                 const float* __restrict__ bias, const float* __restrict__ addsrc,
                                 float* __restrict__ C, int M, int N, int K) {
    extern __shared__ _Float16 ldsW[];        // 64 rows x K halfs, row stride = K

    const int lane   = threadIdx.x & 31;
    const int wave   = threadIdx.x >> 5;
    const int ntiles = N >> 6;
    const int mtb    = blockIdx.x / ntiles;   // block of 8 m-tiles
    const int nt     = blockIdx.x % ntiles;
    const int m0     = (mtb * 8 + wave) * 16;
    const int n0     = nt * 64;

    // ---- TDM: stage W[n0:n0+64, :] into LDS (wave 0 issues one descriptor) ----
    if (threadIdx.x < 32) {
        unsigned long long gaddr = (unsigned long long)(W + (size_t)n0 * K);
        // D# group 0: count=1, lds_addr=0 (only LDS allocation), global_addr, type=2
        v4u g0;
        g0[0] = 1u;                                     // count=1, user mode
        g0[1] = 0u;                                     // lds_addr (bytes)
        g0[2] = (unsigned)(gaddr & 0xffffffffu);        // global_addr[31:0]
        g0[3] = (unsigned)((gaddr >> 32) & 0x1ffffffu)  // global_addr[56:32]
              | (2u << 30);                             // type=2 ("image")
        // D# group 1: data_size=1 (2B), tensor K x 64, tile K x 64, stride0=K
        unsigned td0 = (unsigned)K, td1 = 64u, tl0 = (unsigned)K, tl1 = 64u;
        unsigned long long st0 = (unsigned long long)K;
        v8u g1;
        g1[0] = (1u << 16);                             // wg_mask=0, data_size=1 (2 bytes)
        g1[1] = (td0 & 0xffffu) << 16;                  // tensor_dim0[15:0] @ bits 63:48
        g1[2] = (td0 >> 16) | ((td1 & 0xffffu) << 16);  // tensor_dim0[31:16] | tensor_dim1[15:0]
        g1[3] = (td1 >> 16) | (tl0 << 16);              // tensor_dim1[31:16] | tile_dim0
        g1[4] = tl1;                                    // tile_dim1 | tile_dim2=0
        g1[5] = (unsigned)(st0 & 0xffffffffu);          // tensor_dim0_stride[31:0]
        g1[6] = (unsigned)((st0 >> 32) & 0xffffu);      // stride0[47:32] | stride1[15:0]=0
        g1[7] = 0u;                                     // stride1[47:16]=0
        asm volatile("tensor_load_to_lds %0, %1" :: "s"(g0), "s"(g1) : "memory");
        __builtin_amdgcn_s_wait_tensorcnt(0);
    }
    __syncthreads();

    const int khA = (lane >> 4) * 8;
    const int khB = (lane >> 4) * 8;
    const _Float16* Arow = A + (size_t)(m0 + (lane & 15)) * K;
    const _Float16* Brow = ldsW + (size_t)(lane & 15) * K;

    union UA { v16h v; v8h h[2]; };
    UA cur, nxt;
    cur.h[0] = *(const v8h*)(Arow + khA);
    cur.h[1] = *(const v8h*)(Arow + khA + 16);

    v8f acc0 = {}, acc1 = {}, acc2 = {}, acc3 = {};
    for (int kb = 0; kb < K; kb += 32) {
        if (kb + 32 < K) {   // double-buffer next A fragment (overlaps the WMMAs below)
            __builtin_prefetch(Arow + kb + 64, 0, 1);
            nxt.h[0] = *(const v8h*)(Arow + kb + 32 + khA);
            nxt.h[1] = *(const v8h*)(Arow + kb + 32 + khA + 16);
        }
        v16h a = cur.v;
        UA b0, b1, b2, b3;
        b0.h[0] = *(const v8h*)(Brow + (size_t)0  * K + kb + khB);
        b0.h[1] = *(const v8h*)(Brow + (size_t)0  * K + kb + khB + 16);
        b1.h[0] = *(const v8h*)(Brow + (size_t)16 * K + kb + khB);
        b1.h[1] = *(const v8h*)(Brow + (size_t)16 * K + kb + khB + 16);
        b2.h[0] = *(const v8h*)(Brow + (size_t)32 * K + kb + khB);
        b2.h[1] = *(const v8h*)(Brow + (size_t)32 * K + kb + khB + 16);
        b3.h[0] = *(const v8h*)(Brow + (size_t)48 * K + kb + khB);
        b3.h[1] = *(const v8h*)(Brow + (size_t)48 * K + kb + khB + 16);
        acc0 = __builtin_amdgcn_wmma_f32_16x16x32_f16(false, a, false, b0.v, (short)0, acc0, false, false);
        acc1 = __builtin_amdgcn_wmma_f32_16x16x32_f16(false, a, false, b1.v, (short)0, acc1, false, false);
        acc2 = __builtin_amdgcn_wmma_f32_16x16x32_f16(false, a, false, b2.v, (short)0, acc2, false, false);
        acc3 = __builtin_amdgcn_wmma_f32_16x16x32_f16(false, a, false, b3.v, (short)0, acc3, false, false);
        cur = nxt;
    }

    // C/D layout: lanes 0-15 -> rows v, lanes 16-31 -> rows v+8; col = lane&15 (+16*j)
    const int cn    = n0 + (lane & 15);
    const int mrow0 = m0 + (lane >> 4) * 8;
#pragma unroll
    for (int v = 0; v < 8; ++v) {
        int m = mrow0 + v;
        float vals[4] = { acc0[v], acc1[v], acc2[v], acc3[v] };
#pragma unroll
        for (int j = 0; j < 4; ++j) {
            int col = cn + 16 * j;
            float r = vals[j];
            if (bias)   r += bias[col];
            if (addsrc) r += addsrc[(size_t)m * N + col];
            C[(size_t)m * N + col] = r;
        }
    }
}

// ---------------- causal depthwise conv (DC=4) + SiLU ----------------
__global__ void conv_silu_kernel(const float* __restrict__ xz, const float* __restrict__ cw,
                                 const float* __restrict__ cb, float* __restrict__ xc,
                                 _Float16* __restrict__ xch) {
    int idx = blockIdx.x * blockDim.x + threadIdx.x;
    if (idx >= M_ * DI_) return;
    int d   = idx % DI_;
    int row = idx / DI_;
    int l   = row % L_;
    float acc = cb[d];
#pragma unroll
    for (int j = 0; j < DC_; ++j) {
        int ls = l - (DC_ - 1) + j;
        if (ls >= 0)
            acc += cw[d * DC_ + j] * xz[(size_t)(row - (DC_ - 1) + j) * D2I_ + d];
    }
    float v = silu_f(acc);
    xc[idx]  = v;
    xch[idx] = (_Float16)v;
}

// ---------------- dt = softplus(dbl[:, :12] @ dtproj_w^T + b) ----------------
__global__ void dt_kernel(const float* __restrict__ dbl, const float* __restrict__ dtw,
                          const float* __restrict__ dtb, float* __restrict__ dt) {
    int idx = blockIdx.x * blockDim.x + threadIdx.x;
    if (idx >= M_ * DI_) return;
    int d   = idx % DI_;
    int row = idx / DI_;
    float acc = dtb[d];
#pragma unroll
    for (int j = 0; j < DTR_; ++j)
        acc += dbl[(size_t)row * NPAD_ + j] * dtw[d * DTR_ + j];
    dt[idx] = (acc > 20.f) ? acc : log1pf(expf(acc));
}

// ---------------- selective scan: one lane per (b, d), 16 states in VGPRs ----------------
// Fuses y = (scan + D*xc) * silu(z) and f16 conversion for the out-proj GEMM.
__global__ void scan_kernel(const float* __restrict__ dt, const float* __restrict__ dbl,
                            const float* __restrict__ xc, const float* __restrict__ xz,
                            const float* __restrict__ A_log, const float* __restrict__ Dp,
                            _Float16* __restrict__ y) {
    int b = blockIdx.x;
    int d = threadIdx.x;                      // 0..383
    float Ad[DS_], h[DS_];
#pragma unroll
    for (int s = 0; s < DS_; ++s) { Ad[s] = -expf(A_log[d * DS_ + s]); h[s] = 0.f; }
    float Dv = Dp[d];
    for (int t = 0; t < L_; ++t) {
        size_t row = (size_t)b * L_ + t;
        float dtv = dt[row * DI_ + d];
        float xcv = xc[row * DI_ + d];
        float zv  = xz[row * D2I_ + DI_ + d];
        const float* br = dbl + row * NPAD_;  // cols 12..27 = Bm, 28..43 = C
        float dbx = dtv * xcv;
        float yv = 0.f;
#pragma unroll
        for (int s = 0; s < DS_; ++s) {
            float dA = expf(dtv * Ad[s]);
            h[s] = dA * h[s] + dbx * br[DTR_ + s];
            yv  += h[s] * br[DTR_ + DS_ + s];
        }
        float out = (yv + Dv * xcv) * silu_f(zv);
        y[row * DI_ + d] = (_Float16)out;
    }
}

// ---------------- head: logits = relu(gelu(h1) @ w2 + b2) * mask ----------------
__global__ void head2_kernel(const float* __restrict__ h1, const float* __restrict__ w2,
                             const float* __restrict__ b2, const unsigned char* __restrict__ mask,
                             float* __restrict__ out) {
    int wave = threadIdx.x >> 5, lane = threadIdx.x & 31;
    int row = blockIdx.x * (blockDim.x >> 5) + wave;
    if (row >= M_) return;
    const float* hr = h1 + (size_t)row * DM_;
    float acc = 0.f;
#pragma unroll
    for (int i = 0; i < 6; ++i) {
        int c = lane + 32 * i;
        float v = hr[c];
        float g = 0.5f * v * (1.f + erff(v * 0.70710678118654752f));  // exact gelu
        acc += g * w2[c];
    }
#pragma unroll
    for (int off = 16; off > 0; off >>= 1) acc += __shfl_xor(acc, off, 32);
    if (lane == 0) {
        float lg = acc + b2[0];
        lg = lg > 0.f ? lg : 0.f;
        out[row] = mask[row] ? lg : 0.f;   // jnp bool -> 1 byte per element
    }
}

// ---------------- host side ----------------
static inline void launch_gemm(const _Float16* A, const _Float16* W, const float* bias,
                               const float* addsrc, float* C, int N, int K, hipStream_t s) {
    int blocks = (M_ / 128) * (N / 64);
    size_t shmem = (size_t)64 * K * 2;   // W tile in f16
    wmma_gemm_kernel<<<blocks, 256, shmem, s>>>(A, W, bias, addsrc, C, M_, N, K);
}

extern "C" void kernel_launch(void* const* d_in, const int* in_sizes, int n_in,
                              void* d_out, int out_size, void* d_ws, size_t ws_size,
                              hipStream_t stream) {
    // ---- inputs (setup_inputs order) ----
    const int*           cod_ids   = (const int*)d_in[0];
    const float*         sim_feat  = (const float*)d_in[1];
    const unsigned char* pad_mask  = (const unsigned char*)d_in[2];
    const int*           angle_bin = (const int*)d_in[3];
    const int*           pair_bin  = (const int*)d_in[4];
    const int*           bucket    = (const int*)d_in[5];
    const float*         cod_emb   = (const float*)d_in[6];
    const float*         sim_w     = (const float*)d_in[7];
    const float*         sim_b     = (const float*)d_in[8];
    const float*         angle_w   = (const float*)d_in[9];
    const float*         pair_w    = (const float*)d_in[10];
    const float*         bucket_w  = (const float*)d_in[11];
    const float*         ln_g      = (const float*)d_in[12];
    const float*         ln_b      = (const float*)d_in[13];
    const float*         in_w      = (const float*)d_in[14];
    const float*         in_b      = (const float*)d_in[15];
    const float*         conv_w    = (const float*)d_in[16];
    const float*         conv_b    = (const float*)d_in[17];
    const float*         xproj_w   = (const float*)d_in[18];
    const float*         dtproj_w  = (const float*)d_in[19];
    const float*         dtproj_b  = (const float*)d_in[20];
    const float*         A_log     = (const float*)d_in[21];
    const float*         D_param   = (const float*)d_in[22];
    const float*         out_w     = (const float*)d_in[23];
    const float*         out_b     = (const float*)d_in[24];
    const float*         head_w1   = (const float*)d_in[25];
    const float*         head_b1   = (const float*)d_in[26];
    const float*         head_w2   = (const float*)d_in[27];
    const float*         head_b2   = (const float*)d_in[28];
    float* out = (float*)d_out;
    (void)in_sizes; (void)n_in; (void)out_size; (void)ws_size;

    // ---- workspace carve-out (256B aligned regions) ----
    char* ws = (char*)d_ws;
    size_t off = 0;
    auto alloc = [&](size_t bytes) -> void* {
        void* p = ws + off;
        off += (bytes + 255) & ~(size_t)255;
        return p;
    };
    _Float16* wf_in  = (_Float16*)alloc((size_t)NL_ * D2I_ * DM_ * 2);   // (NL,768,192) f16
    _Float16* wf_xp  = (_Float16*)alloc((size_t)NL_ * NPAD_ * DI_ * 2);  // (NL,64,384) f16 padded
    _Float16* wf_out = (_Float16*)alloc((size_t)NL_ * DM_ * DI_ * 2);    // (NL,192,384) f16
    _Float16* wf_h1  = (_Float16*)alloc((size_t)DM_ * DM_ * 2);          // (192,192) f16
    float*    x      = (float*)alloc((size_t)M_ * DM_ * 4);
    _Float16* hf16   = (_Float16*)alloc((size_t)M_ * DM_ * 2);
    float*    xz     = (float*)alloc((size_t)M_ * D2I_ * 4);
    float*    xcf    = (float*)alloc((size_t)M_ * DI_ * 4);
    _Float16* xch    = (_Float16*)alloc((size_t)M_ * DI_ * 2);
    float*    dbl    = (float*)alloc((size_t)M_ * NPAD_ * 4);
    float*    dtb    = (float*)alloc((size_t)M_ * DI_ * 4);
    _Float16* yf16   = (_Float16*)alloc((size_t)M_ * DI_ * 2);
    float*    h1     = (float*)alloc((size_t)M_ * DM_ * 4);

    const int TPB = 256;

    // ---- weight prep ----
    f32tof16_kernel<<<(NL_ * D2I_ * DM_) / TPB, TPB, 0, stream>>>(in_w,  wf_in,  NL_ * D2I_ * DM_);
    f32tof16_kernel<<<(NL_ * DM_ * DI_) / TPB, TPB, 0, stream>>>(out_w, wf_out, NL_ * DM_ * DI_);
    f32tof16_kernel<<<(DM_ * DM_) / TPB, TPB, 0, stream>>>(head_w1, wf_h1, DM_ * DM_);
    pad_xproj_kernel<<<(NL_ * NPAD_ * DI_) / TPB, TPB, 0, stream>>>(xproj_w, wf_xp);

    // ---- embedding ----
    embed_kernel<<<(M_ * DM_) / TPB, TPB, 0, stream>>>(cod_ids, sim_feat, angle_bin, pair_bin,
                                                       bucket, cod_emb, sim_w, sim_b,
                                                       angle_w, pair_w, bucket_w, x);

    // ---- 4 Mamba layers ----
    for (int i = 0; i < NL_; ++i) {
        ln_kernel<<<M_ / 8, TPB, 0, stream>>>(x, ln_g + i * DM_, ln_b + i * DM_, hf16);
        launch_gemm(hf16, wf_in + (size_t)i * D2I_ * DM_, in_b + i * D2I_, nullptr,
                    xz, D2I_, DM_, stream);
        conv_silu_kernel<<<(M_ * DI_) / TPB, TPB, 0, stream>>>(
            xz, conv_w + (size_t)i * DI_ * DC_, conv_b + i * DI_, xcf, xch);
        launch_gemm(xch, wf_xp + (size_t)i * NPAD_ * DI_, nullptr, nullptr,
                    dbl, NPAD_, DI_, stream);
        dt_kernel<<<(M_ * DI_) / TPB, TPB, 0, stream>>>(
            dbl, dtproj_w + (size_t)i * DI_ * DTR_, dtproj_b + i * DI_, dtb);
        scan_kernel<<<B_, DI_, 0, stream>>>(
            dtb, dbl, xcf, xz, A_log + (size_t)i * DI_ * DS_, D_param + i * DI_, yf16);
        launch_gemm(yf16, wf_out + (size_t)i * DM_ * DI_, out_b + i * DM_, x,
                    x, DM_, DI_, stream);   // fused residual add, in-place
    }

    // ---- head ----
    f32tof16_kernel<<<(M_ * DM_) / TPB, TPB, 0, stream>>>(x, hf16, M_ * DM_);
    launch_gemm(hf16, wf_h1, head_b1, nullptr, h1, DM_, DM_, stream);
    head2_kernel<<<M_ / 8, TPB, 0, stream>>>(h1, head_w2, head_b2, pad_mask, out);
}
